// DQN_86655260164180
// MI455X (gfx1250) — matrix operand order repaired and checked by reading, not compiled, not run on hardware
//
#include <hip/hip_runtime.h>

#define NN 50000
#define EE 400000
#define NH 4
#define ETOT (EE + NN)
#define LEPS 1e-5f

typedef _Float16 v16h __attribute__((ext_vector_type(16)));
typedef _Float16 v8h  __attribute__((ext_vector_type(8)));
typedef float    v8f  __attribute__((ext_vector_type(8)));

#define CDIV(a,b) (((a)+(b)-1)/(b))

// ---------------------------------------------------------------- utilities
__device__ __forceinline__ unsigned hash_u32(unsigned a) {
  a ^= a >> 16; a *= 0x7feb352dU; a ^= a >> 15; a *= 0x846ca68bU; a ^= a >> 16;
  return a;
}

__device__ __forceinline__ void atomic_max_f32(float* addr, float val) {
  int* ai = (int*)addr;
  int old = __float_as_int(*addr);
  while (__int_as_float(old) < val) {
    int assumed = old;
    old = atomicCAS(ai, assumed, __float_as_int(val));
    if (old == assumed) break;
  }
}

__device__ __forceinline__ void edge_sd(const int* __restrict__ ei, long long e,
                                        int& s, int& d) {
  if (e < EE) { s = ei[e]; d = ei[EE + e]; }
  else        { s = d = (int)(e - EE); }
}

__global__ void fill_f32(float* p, float v, long long n) {
  long long i = (long long)blockIdx.x * blockDim.x + threadIdx.x;
  if (i < n) p[i] = v;
}
__global__ void fill_i32(int* p, int v, long long n) {
  long long i = (long long)blockIdx.x * blockDim.x + threadIdx.x;
  if (i < n) p[i] = v;
}

// in_x = x[:, 1:5]  (N,6) -> (N,32) fp16, zero-padded cols 4..31
__global__ void extract_x_f16(const float* __restrict__ x, _Float16* __restrict__ d) {
  int t = blockIdx.x * blockDim.x + threadIdx.x;
  if (t >= NN * 32) return;
  int r = t >> 5, c = t & 31;
  d[t] = (c < 4) ? (_Float16)x[r * 6 + 1 + c] : (_Float16)0.f;
}

// ---------------------------------------------------- weight pack (B fragments)
// W (f32, Kreal x Ncols, row-major) -> fp16 fragments, K zero-padded to Kpad.
// Packed layout: bp[((kc*nTn + tn)*32 + lane)*16 + j], where per-lane element j
// holds B[kc*32 + (lane>>4)*16 + j, tn*16 + (lane&15)]  (contiguous j == contiguous K).
__global__ void pack_b_f16(const float* __restrict__ W, _Float16* __restrict__ bp,
                           int Kreal, int Kpad, int Ncols) {
  int t = blockIdx.x * blockDim.x + threadIdx.x;
  if (t >= Kpad * Ncols) return;
  int j    = t & 15;
  int lane = (t >> 4) & 31;
  int rest = t >> 9;                 // kc*nTn + tn
  int nTn  = Ncols >> 4;
  int tn   = rest % nTn;
  int kc   = rest / nTn;
  int kb   = kc * 32 + (lane >> 4) * 16 + j;
  int col  = tn * 16 + (lane & 15);
  bp[t] = (kb < Kreal) ? (_Float16)W[(size_t)kb * Ncols + col] : (_Float16)0.f;
}

// ------------------------------------------------------- WMMA f16 GEMM (f32 acc)
// C[M,Ncols] = A[M,K] * B[K,Ncols]; A row-major fp16, B pre-packed fragments.
// Requires: K%32==0, M%16==0, (Ncols/16)%4==0. One wave per 16x16 C tile.
__global__ __launch_bounds__(128)
void wmma_gemm_f16(const _Float16* __restrict__ A, const _Float16* __restrict__ Bp,
                   float* __restrict__ C, int M, int Ncols, int K) {
  int lane  = threadIdx.x & 31;
  int wave  = threadIdx.x >> 5;
  int tn    = blockIdx.x * 4 + wave;
  int tm    = blockIdx.y;
  int group = lane >> 4;
  int l16   = lane & 15;
  int nTn   = Ncols >> 4;

  const _Float16* Arow = A + (size_t)(tm * 16 + l16) * K + group * 16;
  const _Float16* Brow = Bp + ((size_t)tn * 32 + lane) * 16;
  size_t bstep = (size_t)nTn * 512;   // halves per k-chunk of packed B

  v8f acc = {};
  for (int kc = 0; kc < (K >> 5); ++kc) {
    v8h a0 = *(const v8h*)(Arow);        // K = kc*32 + group*16 .. +7
    v8h a1 = *(const v8h*)(Arow + 8);    // K = kc*32 + group*16 + 8 .. +15
    v16h b = *(const v16h*)(Brow);
    v16h a;
#pragma unroll
    for (int i = 0; i < 8; ++i) { a[i] = a0[i]; a[i + 8] = a1[i]; }
    acc = __builtin_amdgcn_wmma_f32_16x16x32_f16(false, a, false, b,
                                                 (short)0, acc, false, false);
    Arow += 32;
    Brow += bstep;
  }
  float* Crow = C + (size_t)(tm * 16 + 8 * group) * Ncols + tn * 16 + l16;
#pragma unroll
  for (int v = 0; v < 8; ++v) Crow[(size_t)v * Ncols] = acc[v];
}

// ------------------------------------------------------------- GAT edge kernels
__global__ void attn_scores(const float* __restrict__ hmat, const float* __restrict__ a_s,
                            const float* __restrict__ a_d, float* __restrict__ lsb,
                            float* __restrict__ ldb, int C) {
  int t = blockIdx.x * blockDim.x + threadIdx.x;
  if (t >= NN * NH) return;
  int n = t / NH, h = t % NH;
  const float* hp = hmat + (size_t)n * NH * C + (size_t)h * C;
  const float* sp = a_s + h * C;
  const float* dp = a_d + h * C;
  float s = 0.f, d = 0.f;
  for (int c = 0; c < C; ++c) { float v = hp[c]; s += v * sp[c]; d += v * dp[c]; }
  lsb[t] = s; ldb[t] = d;
}

__global__ void edge_logit_max(const int* __restrict__ ei, const float4* __restrict__ lsb,
                               const float4* __restrict__ ldb, float4* __restrict__ logit,
                               float* __restrict__ mmax) {
  long long e = (long long)blockIdx.x * blockDim.x + threadIdx.x;
  if (e >= ETOT) return;
  int s, d; edge_sd(ei, e, s, d);
  float4 a = lsb[s], b = ldb[d];
  float v0 = a.x + b.x, v1 = a.y + b.y, v2 = a.z + b.z, v3 = a.w + b.w;
  v0 = (v0 >= 0.f) ? v0 : 0.2f * v0;
  v1 = (v1 >= 0.f) ? v1 : 0.2f * v1;
  v2 = (v2 >= 0.f) ? v2 : 0.2f * v2;
  v3 = (v3 >= 0.f) ? v3 : 0.2f * v3;
  logit[e] = make_float4(v0, v1, v2, v3);
  atomic_max_f32(&mmax[d * NH + 0], v0);
  atomic_max_f32(&mmax[d * NH + 1], v1);
  atomic_max_f32(&mmax[d * NH + 2], v2);
  atomic_max_f32(&mmax[d * NH + 3], v3);
}

__global__ void edge_exp_sum(const int* __restrict__ ei, float4* __restrict__ logit,
                             const float4* __restrict__ mmax, float* __restrict__ den) {
  long long e = (long long)blockIdx.x * blockDim.x + threadIdx.x;
  if (e >= ETOT) return;
  int s, d; edge_sd(ei, e, s, d);
  float4 lg = logit[e], m = mmax[d];
  float e0 = __expf(lg.x - m.x), e1 = __expf(lg.y - m.y);
  float e2 = __expf(lg.z - m.z), e3 = __expf(lg.w - m.w);
  logit[e] = make_float4(e0, e1, e2, e3);
  atomicAdd(&den[d * NH + 0], e0);
  atomicAdd(&den[d * NH + 1], e1);
  atomicAdd(&den[d * NH + 2], e2);
  atomicAdd(&den[d * NH + 3], e3);
}

// agg[dst,c] += sum_h h[src,h,c] * alpha[e,h]   (one atomic per (e,c))
__global__ void edge_aggregate(const int* __restrict__ ei, const float* __restrict__ hmat,
                               const float4* __restrict__ exb, const float4* __restrict__ den,
                               float* __restrict__ agg, int C) {
  long long t = (long long)blockIdx.x * blockDim.x + threadIdx.x;
  if (t >= (long long)ETOT * C) return;
  long long e = t / C;
  int c = (int)(t % C);
  int s, d; edge_sd(ei, e, s, d);
  float4 ex = exb[e], dn = den[d];
  const float* hp = hmat + (size_t)s * NH * C + c;
  float acc = hp[0 * C] * (ex.x / dn.x) + hp[1 * C] * (ex.y / dn.y) +
              hp[2 * C] * (ex.z / dn.z) + hp[3 * C] * (ex.w / dn.w);
  atomicAdd(&agg[(size_t)d * C + c], acc);
}

// --------------------------------------------------------------- batch norm
__global__ __launch_bounds__(256)
void bn_stats(const float* __restrict__ agg, const float* __restrict__ bias, float scale,
              float* __restrict__ sums, float* __restrict__ sumsq, int C) {
  __shared__ float sm_s[256], sm_q[256];
  int c   = threadIdx.x % C;
  int rpb = 256 / C;
  int n0  = blockIdx.x * rpb + threadIdx.x / C;
  int stride = gridDim.x * rpb;
  float s = 0.f, q = 0.f;
  for (int n = n0; n < NN; n += stride) {
    float v = agg[(size_t)n * C + c] * scale + bias[c];
    s += v; q += v * v;
  }
  sm_s[threadIdx.x] = s; sm_q[threadIdx.x] = q;
  __syncthreads();
  if (threadIdx.x < C) {
    float S = sm_s[threadIdx.x], Q = sm_q[threadIdx.x];
    for (int i = threadIdx.x + C; i < 256; i += C) { S += sm_s[i]; Q += sm_q[i]; }
    atomicAdd(&sums[threadIdx.x], S);
    atomicAdd(&sumsq[threadIdx.x], Q);
  }
}

__global__ void bn_apply_drop(const float* __restrict__ agg, const float* __restrict__ bias,
                              float scale, const float* __restrict__ sums,
                              const float* __restrict__ sumsq, const float* __restrict__ g,
                              const float* __restrict__ be, _Float16* __restrict__ out16,
                              int C, unsigned seed, int doElu) {
  long long t = (long long)blockIdx.x * blockDim.x + threadIdx.x;
  if (t >= (long long)NN * C) return;
  int c = (int)(t % C);
  float mean = sums[c] / (float)NN;
  float var  = sumsq[c] / (float)NN - mean * mean;
  float v = agg[t] * scale + bias[c];
  v = (v - mean) * rsqrtf(var + LEPS) * g[c] + be[c];
  if (doElu) v = (v > 0.f) ? v : (__expf(v) - 1.f);
  unsigned hsh = hash_u32((unsigned)t * 2654435761u ^ seed);
  v = (hsh & 1u) ? v * 2.f : 0.f;                    // dropout p=0.5 (det. hash)
  out16[t] = (_Float16)v;
}

// ------------------------------------------- ordered compaction of cond nodes
__global__ __launch_bounds__(1024)
void build_idx(const float* __restrict__ x, int* __restrict__ idx) {
  __shared__ int sc[1024];
  __shared__ int base;
  if (threadIdx.x == 0) base = 0;
  __syncthreads();
  for (int start = 0; start < NN; start += 1024) {
    int n = start + threadIdx.x;
    int flag = 0;
    if (n < NN) flag = (x[n * 6 + 2] == 1.0f) && (x[n * 6 + 5] == 0.0f);
    sc[threadIdx.x] = flag;
    __syncthreads();
    for (int off = 1; off < 1024; off <<= 1) {
      int v = (threadIdx.x >= off) ? sc[threadIdx.x - off] : 0;
      __syncthreads();
      sc[threadIdx.x] += v;
      __syncthreads();
    }
    if (flag) idx[base + sc[threadIdx.x] - 1] = n;
    __syncthreads();
    if (threadIdx.x == 0) base += sc[1023];
    __syncthreads();
  }
}

__global__ void gather_rows_f16(const _Float16* __restrict__ src, const int* __restrict__ idx,
                                _Float16* __restrict__ dst, int C) {
  long long t = (long long)blockIdx.x * blockDim.x + threadIdx.x;
  if (t >= (long long)NN * C) return;
  int n = (int)(t / C), c = (int)(t % C);
  dst[t] = src[(size_t)idx[n] * C + c];
}

// ----------------------------------------------------------- MLP head pieces
__global__ void ln_act(const float* __restrict__ in, const float* __restrict__ bias,
                       const float* __restrict__ g, const float* __restrict__ b,
                       _Float16* __restrict__ out16, float* __restrict__ out32, int C) {
  int n = blockIdx.x * blockDim.x + threadIdx.x;
  if (n >= NN) return;
  const float* r = in + (size_t)n * C;
  float mu = 0.f;
  for (int c = 0; c < C; ++c) mu += r[c] + bias[c];
  mu /= (float)C;
  float var = 0.f;
  for (int c = 0; c < C; ++c) { float d = r[c] + bias[c] - mu; var += d * d; }
  var /= (float)C;
  float rs = rsqrtf(var + LEPS);
  for (int c = 0; c < C; ++c) {
    float v = (r[c] + bias[c] - mu) * rs * g[c] + b[c];
    v = fmaxf(v, 0.f);
    out16[(size_t)n * C + c] = (_Float16)v;
    if (out32) out32[(size_t)n * C + c] = v;
  }
}

__global__ void head_dot_tanh(const float* __restrict__ a2, const float* __restrict__ A3,
                              const float* __restrict__ ab3, float* __restrict__ logits,
                              int C) {
  int n = blockIdx.x * blockDim.x + threadIdx.x;
  if (n >= NN) return;
  float s = ab3[0];
  for (int c = 0; c < C; ++c) s += a2[(size_t)n * C + c] * A3[c];
  logits[n] = tanhf(s);
}

__global__ void reduce_max_k(const float* __restrict__ v, int n, float* __restrict__ out) {
  __shared__ float sm[256];
  float m = -3.0e38f;
  for (int i = blockIdx.x * 256 + threadIdx.x; i < n; i += gridDim.x * 256)
    m = fmaxf(m, v[i]);
  sm[threadIdx.x] = m; __syncthreads();
  for (int off = 128; off; off >>= 1) {
    if (threadIdx.x < off) sm[threadIdx.x] = fmaxf(sm[threadIdx.x], sm[threadIdx.x + off]);
    __syncthreads();
  }
  if (threadIdx.x == 0) atomic_max_f32(out, sm[0]);
}

__global__ void reduce_sumexp_k(const float* __restrict__ v, int n,
                                const float* __restrict__ mx, float* __restrict__ out) {
  __shared__ float sm[256];
  float s = 0.f;
  float m = *mx;
  for (int i = blockIdx.x * 256 + threadIdx.x; i < n; i += gridDim.x * 256)
    s += __expf(v[i] - m);
  sm[threadIdx.x] = s; __syncthreads();
  for (int off = 128; off; off >>= 1) {
    if (threadIdx.x < off) sm[threadIdx.x] += sm[threadIdx.x + off];
    __syncthreads();
  }
  if (threadIdx.x == 0) atomicAdd(out, sm[0]);
}

__global__ void write_prob_k(const float* __restrict__ v, int n, const float* __restrict__ mx,
                             const float* __restrict__ sum, float* __restrict__ prob) {
  int i = blockIdx.x * blockDim.x + threadIdx.x;
  if (i < n) prob[i] = __expf(v[i] - *mx) / (*sum);
}

// ================================================================= launcher
struct Ws {
  float* hbuf;       // NN*512 f32 (also reused for MLP intermediates)
  float* agg;        // NN*128 f32
  _Float16* buf16a;  // NN*128 f16
  _Float16* buf16b;  // NN*128 f16
  _Float16* w16;     // 32768 f16 (packed B fragments)
  float *lsb, *ldb, *mmax, *den;   // NN*NH each
  float* exb;        // ETOT*NH
  float* stats;      // 256
  int* idx;          // NN
};

static void gat_layer(const _Float16* in16, int Kreal, int Kpad, int C, const float* W,
                      const float* a_s, const float* a_d, const float* bias,
                      const float* g, const float* be, int doElu, unsigned seed,
                      const int* ei, const Ws& w, _Float16* out16, hipStream_t st) {
  int Ncols = NH * C;
  pack_b_f16<<<CDIV(Kpad * Ncols, 256), 256, 0, st>>>(W, w.w16, Kreal, Kpad, Ncols);
  dim3 gg(Ncols / 64, NN / 16);
  wmma_gemm_f16<<<gg, 128, 0, st>>>(in16, w.w16, w.hbuf, NN, Ncols, Kpad);
  attn_scores<<<CDIV(NN * NH, 256), 256, 0, st>>>(w.hbuf, a_s, a_d, w.lsb, w.ldb, C);
  fill_f32<<<CDIV(NN * NH, 256), 256, 0, st>>>(w.mmax, -3.0e38f, NN * NH);
  fill_f32<<<CDIV(NN * NH, 256), 256, 0, st>>>(w.den, 0.f, NN * NH);
  edge_logit_max<<<CDIV(ETOT, 256), 256, 0, st>>>(ei, (const float4*)w.lsb,
                                                  (const float4*)w.ldb, (float4*)w.exb,
                                                  w.mmax);
  edge_exp_sum<<<CDIV(ETOT, 256), 256, 0, st>>>(ei, (float4*)w.exb,
                                                (const float4*)w.mmax, w.den);
  fill_f32<<<CDIV((long long)NN * C, 256), 256, 0, st>>>(w.agg, 0.f, (long long)NN * C);
  edge_aggregate<<<CDIV((long long)ETOT * C, 256), 256, 0, st>>>(
      ei, w.hbuf, (const float4*)w.exb, (const float4*)w.den, w.agg, C);
  fill_f32<<<1, 256, 0, st>>>(w.stats, 0.f, 2 * C);
  bn_stats<<<256, 256, 0, st>>>(w.agg, bias, 1.f / NH, w.stats, w.stats + C, C);
  bn_apply_drop<<<CDIV((long long)NN * C, 256), 256, 0, st>>>(
      w.agg, bias, 1.f / NH, w.stats, w.stats + C, g, be, out16, C, seed, doElu);
}

extern "C" void kernel_launch(void* const* d_in, const int* in_sizes, int n_in,
                              void* d_out, int out_size, void* d_ws, size_t ws_size,
                              hipStream_t stream) {
  const float* x   = (const float*)d_in[0];
  const int*   ei  = (const int*)d_in[1];
  const float* W1  = (const float*)d_in[2];
  const float* as1 = (const float*)d_in[3];
  const float* ad1 = (const float*)d_in[4];
  const float* b1  = (const float*)d_in[5];
  const float* g1  = (const float*)d_in[6];
  const float* be1 = (const float*)d_in[7];
  const float* W2  = (const float*)d_in[8];
  const float* as2 = (const float*)d_in[9];
  const float* ad2 = (const float*)d_in[10];
  const float* b2  = (const float*)d_in[11];
  const float* g2  = (const float*)d_in[12];
  const float* be2 = (const float*)d_in[13];
  const float* W3  = (const float*)d_in[14];
  const float* as3 = (const float*)d_in[15];
  const float* ad3 = (const float*)d_in[16];
  const float* b3  = (const float*)d_in[17];
  const float* g3  = (const float*)d_in[18];
  const float* be3 = (const float*)d_in[19];
  const float* A1  = (const float*)d_in[20];
  const float* ab1 = (const float*)d_in[21];
  const float* l1g = (const float*)d_in[22];
  const float* l1b = (const float*)d_in[23];
  const float* A2  = (const float*)d_in[24];
  const float* ab2 = (const float*)d_in[25];
  const float* l2g = (const float*)d_in[26];
  const float* l2b = (const float*)d_in[27];
  const float* A3  = (const float*)d_in[28];
  const float* ab3 = (const float*)d_in[29];

  char* p = (char*)d_ws;
  Ws w;
  w.hbuf   = (float*)p;                 p += (size_t)NN * 512 * 4;
  w.agg    = (float*)p;                 p += (size_t)NN * 128 * 4;
  w.buf16a = (_Float16*)p;              p += (size_t)NN * 128 * 2;
  w.buf16b = (_Float16*)p;              p += (size_t)NN * 128 * 2;
  w.w16    = (_Float16*)p;              p += 32768 * 2;
  w.lsb    = (float*)p;                 p += (size_t)NN * NH * 4;
  w.ldb    = (float*)p;                 p += (size_t)NN * NH * 4;
  w.mmax   = (float*)p;                 p += (size_t)NN * NH * 4;
  w.den    = (float*)p;                 p += (size_t)NN * NH * 4;
  w.exb    = (float*)p;                 p += (size_t)ETOT * NH * 4;
  w.stats  = (float*)p;                 p += 256 * 4;
  float* scal = (float*)p;              p += 16 * 4;
  w.idx    = (int*)p;                   p += (size_t)NN * 4;

  float* logits = (float*)d_out;
  float* prob   = logits + NN;

  // ---- GAT layer 1 (in: x[:,1:5] zero-padded to K=32, C=128, elu+drop)
  extract_x_f16<<<CDIV(NN * 32, 256), 256, 0, stream>>>(x, w.buf16a);
  gat_layer(w.buf16a, 4, 32, 128, W1, as1, ad1, b1, g1, be1, 1, 0x9E3779B9u, ei, w,
            w.buf16b, stream);
  // ---- GAT layer 2 (K=128, C=64, elu+drop)
  gat_layer(w.buf16b, 128, 128, 64, W2, as2, ad2, b2, g2, be2, 1, 0x3C6EF372u, ei, w,
            w.buf16a, stream);
  // ---- GAT layer 3 (K=64, C=32, no elu, drop)
  gat_layer(w.buf16a, 64, 64, 32, W3, as3, ad3, b3, g3, be3, 0, 0xDAA66D2Bu, ei, w,
            w.buf16b, stream);

  // ---- selection: idx = nonzero(cond, size=N, fill=0); sel = h[idx]
  fill_i32<<<CDIV(NN, 256), 256, 0, stream>>>(w.idx, 0, NN);
  build_idx<<<1, 1024, 0, stream>>>(x, w.idx);
  gather_rows_f16<<<CDIV((long long)NN * 32, 256), 256, 0, stream>>>(w.buf16b, w.idx,
                                                                     w.buf16a, 32);

  // ---- MLP head: relu(ln(sel@A1+ab1)) -> relu(ln(.@A2+ab2)) -> .@A3+ab3 -> tanh
  float* mlpA = w.hbuf;                         // NN*128
  float* mlpB = w.hbuf + (size_t)NN * 128;      // NN*64
  float* mlpC = w.hbuf + (size_t)NN * 192;      // NN*64

  pack_b_f16<<<CDIV(32 * 128, 256), 256, 0, stream>>>(A1, w.w16, 32, 32, 128);
  { dim3 gg(128 / 64, NN / 16);
    wmma_gemm_f16<<<gg, 128, 0, stream>>>(w.buf16a, w.w16, mlpA, NN, 128, 32); }
  ln_act<<<CDIV(NN, 256), 256, 0, stream>>>(mlpA, ab1, l1g, l1b, w.buf16b, nullptr, 128);

  pack_b_f16<<<CDIV(128 * 64, 256), 256, 0, stream>>>(A2, w.w16, 128, 128, 64);
  { dim3 gg(64 / 64, NN / 16);
    wmma_gemm_f16<<<gg, 128, 0, stream>>>(w.buf16b, w.w16, mlpB, NN, 64, 128); }
  ln_act<<<CDIV(NN, 256), 256, 0, stream>>>(mlpB, ab2, l2g, l2b, w.buf16a, mlpC, 64);

  head_dot_tanh<<<CDIV(NN, 256), 256, 0, stream>>>(mlpC, A3, ab3, logits, 64);

  // ---- softmax over all N logits
  fill_f32<<<1, 32, 0, stream>>>(scal, -3.0e38f, 1);
  fill_f32<<<1, 32, 0, stream>>>(scal + 1, 0.f, 1);
  reduce_max_k<<<256, 256, 0, stream>>>(logits, NN, scal);
  reduce_sumexp_k<<<256, 256, 0, stream>>>(logits, NN, scal, scal + 1);
  write_prob_k<<<CDIV(NN, 256), 256, 0, stream>>>(logits, NN, scal, scal + 1, prob);

  (void)in_sizes; (void)n_in; (void)out_size; (void)ws_size;
}
